// MoxELayer_35734127902862
// MI455X (gfx1250) — matrix operand order repaired and checked by reading, not compiled, not run on hardware
//
#include <hip/hip_runtime.h>
#include <hip/hip_bf16.h>

// ---------------- problem constants ----------------
#define B_  4
#define S_  2048
#define D_  1024
#define E_  8
#define F_  4096
#define T_  (B_ * S_)            // 8192 tokens
#define TOPK 2
#define NSLOT (T_ * TOPK)        // 16384 assignments
#define SLOT_CAP (NSLOT + E_ * 128)  // 17408 (128-padded per-expert regions)
#define MTILES (SLOT_CAP / 128)      // 136

typedef unsigned short ushort_t;
typedef __attribute__((ext_vector_type(16))) __bf16 v16bf;
typedef __attribute__((ext_vector_type(8)))  float  v8f;

union FragU { v16bf v; uint4 q[2]; };
union PackU { uint4 q; ushort_t s[8]; };

__device__ __forceinline__ ushort_t f2bf(float f) {
    unsigned int u = __float_as_uint(f);
    unsigned int r = u + 0x7fffu + ((u >> 16) & 1u);
    return (ushort_t)(r >> 16);
}

__device__ __forceinline__ float gelu_tanh(float v) {
    float c = 0.7978845608028654f * (v + 0.044715f * v * v * v);
    return 0.5f * v * (1.0f + tanhf(c));
}

// ---------------- K0: zero / init ----------------
__global__ __launch_bounds__(256) void moxe_init(
    float* __restrict__ fhs, int* __restrict__ slot_token,
    int* __restrict__ cnt, int* __restrict__ fill,
    float* __restrict__ probsum, float* __restrict__ zsum)
{
    size_t i = (size_t)blockIdx.x * 256 + threadIdx.x;
    if (i < (size_t)T_ * D_) fhs[i] = 0.0f;
    if (i < SLOT_CAP) slot_token[i] = -1;
    if (i < E_) { cnt[i] = 0; fill[i] = 0; probsum[i] = 0.0f; }
    if (i == 0) zsum[0] = 0.0f;
}

// ---------------- K4: fp32 -> bf16 for w1, w2 and x (vectorized) ----------------
__global__ __launch_bounds__(256) void moxe_cvt(
    const float* __restrict__ w1, const float* __restrict__ w2,
    const float* __restrict__ x,
    ushort_t* __restrict__ w1b, ushort_t* __restrict__ w2b,
    ushort_t* __restrict__ xb)
{
    size_t n8 = (size_t)E_ * D_ * F_ / 8;
    size_t nx8 = (size_t)T_ * D_ / 8;
    size_t stride = (size_t)gridDim.x * 256;
    for (size_t i = (size_t)blockIdx.x * 256 + threadIdx.x; i < n8; i += stride) {
        const float4* p1 = (const float4*)(w1 + i * 8);
        float4 a = p1[0], b = p1[1];
        PackU u;
        u.s[0] = f2bf(a.x); u.s[1] = f2bf(a.y); u.s[2] = f2bf(a.z); u.s[3] = f2bf(a.w);
        u.s[4] = f2bf(b.x); u.s[5] = f2bf(b.y); u.s[6] = f2bf(b.z); u.s[7] = f2bf(b.w);
        ((uint4*)w1b)[i] = u.q;
        const float4* p2 = (const float4*)(w2 + i * 8);
        float4 c = p2[0], d = p2[1];
        PackU v;
        v.s[0] = f2bf(c.x); v.s[1] = f2bf(c.y); v.s[2] = f2bf(c.z); v.s[3] = f2bf(c.w);
        v.s[4] = f2bf(d.x); v.s[5] = f2bf(d.y); v.s[6] = f2bf(d.z); v.s[7] = f2bf(d.w);
        ((uint4*)w2b)[i] = v.q;
        if (i < nx8) {
            const float4* p3 = (const float4*)(x + i * 8);
            float4 g = p3[0], h = p3[1];
            PackU w;
            w.s[0] = f2bf(g.x); w.s[1] = f2bf(g.y); w.s[2] = f2bf(g.z); w.s[3] = f2bf(g.w);
            w.s[4] = f2bf(h.x); w.s[5] = f2bf(h.y); w.s[6] = f2bf(h.z); w.s[7] = f2bf(h.w);
            ((uint4*)xb)[i] = w.q;
        }
    }
}

// ---------------- K1: router (one wave per token) ----------------
__global__ __launch_bounds__(256) void moxe_router(
    const float* __restrict__ x, const float* __restrict__ wg,
    const float* __restrict__ bg,
    float* __restrict__ gate_logits, float* __restrict__ probs_out,
    int* __restrict__ top_idx, float* __restrict__ top_w,
    int* __restrict__ cnt, float* __restrict__ probsum, float* __restrict__ zsum)
{
    int wave = threadIdx.x >> 5, lane = threadIdx.x & 31;
    int t = blockIdx.x * 8 + wave;
    const float* xr = x + (size_t)t * D_;
    float acc[E_];
#pragma unroll
    for (int e = 0; e < E_; ++e) acc[e] = 0.0f;
    for (int d = lane; d < D_; d += 32) {
        float xv = xr[d];
        const float* wr = wg + (size_t)d * E_;
#pragma unroll
        for (int e = 0; e < E_; ++e) acc[e] += xv * wr[e];
    }
#pragma unroll
    for (int off = 16; off > 0; off >>= 1) {
#pragma unroll
        for (int e = 0; e < E_; ++e) acc[e] += __shfl_xor(acc[e], off, 32);
    }
    if (lane == 0) {
        float l[E_], p[E_];
        float mx = -3.4e38f;
#pragma unroll
        for (int e = 0; e < E_; ++e) { l[e] = acc[e] + bg[e]; mx = fmaxf(mx, l[e]); }
        float s = 0.0f;
#pragma unroll
        for (int e = 0; e < E_; ++e) { p[e] = __expf(l[e] - mx); s += p[e]; }
        float inv = 1.0f / s;
#pragma unroll
        for (int e = 0; e < E_; ++e) {
            p[e] *= inv;
            gate_logits[(size_t)t * E_ + e] = l[e];
            probs_out[(size_t)t * E_ + e]  = p[e];
            atomicAdd(&probsum[e], p[e]);
        }
        float lse = mx + __logf(s);
        atomicAdd(zsum, lse * lse);
        int i1 = 0; float p1 = p[0];
#pragma unroll
        for (int e = 1; e < E_; ++e) if (p[e] > p1) { p1 = p[e]; i1 = e; }
        int i2 = -1; float p2 = -3.4e38f;
#pragma unroll
        for (int e = 0; e < E_; ++e) if (e != i1 && p[e] > p2) { p2 = p[e]; i2 = e; }
        float rs = 1.0f / (p1 + p2);
        top_idx[t * 2 + 0] = i1;  top_w[t * 2 + 0] = p1 * rs;
        top_idx[t * 2 + 1] = i2;  top_w[t * 2 + 1] = p2 * rs;
        atomicAdd(&cnt[i1], 1);
        atomicAdd(&cnt[i2], 1);
    }
}

// ---------------- K2: finalize scalars + padded prefix sums ----------------
__global__ void moxe_finalize(
    const int* __restrict__ cnt, const float* __restrict__ probsum,
    const float* __restrict__ zsum, int* __restrict__ bp,
    float* __restrict__ out_z, float* __restrict__ out_lb,
    float* __restrict__ out_eload, float* __restrict__ out_ecnt)
{
    if (threadIdx.x == 0 && blockIdx.x == 0) {
        bp[0] = 0;
        for (int e = 0; e < E_; ++e) bp[e + 1] = bp[e] + ((cnt[e] + 127) & ~127);
        out_z[0] = zsum[0] / (float)T_;
        float lb = 0.0f;
        for (int e = 0; e < E_; ++e) {
            float frac = (float)cnt[e] / (float)(T_ * TOPK);
            float el = probsum[e] / (float)T_;
            lb += frac * el;
            out_eload[e] = el;
            out_ecnt[e] = (float)cnt[e];
        }
        out_lb[0] = (float)E_ * lb;
    }
}

// ---------------- K3: slot assignment (expert-contiguous) ----------------
__global__ __launch_bounds__(256) void moxe_assign(
    const int* __restrict__ top_idx, const float* __restrict__ top_w,
    const int* __restrict__ bp, int* __restrict__ fill,
    int* __restrict__ slot_token, float* __restrict__ slot_weight)
{
    int t = blockIdx.x * 256 + threadIdx.x;
    if (t >= T_) return;
#pragma unroll
    for (int k = 0; k < TOPK; ++k) {
        int e = top_idx[t * 2 + k];
        int p = atomicAdd(&fill[e], 1);
        int pos = bp[e] + p;
        slot_token[pos]  = t;
        slot_weight[pos] = top_w[t * 2 + k];
    }
}

// ---------------- K5: grouped GEMM1 + GELU -> H1 (bf16) ----------------
// 128x128 tile, K-step 32, 8 waves, ping-pong LDS (1 barrier/step),
// pipelined b128 loads, staged coalesced bf16 epilogue.
__global__ __launch_bounds__(256) void moxe_ffn1(
    const ushort_t* __restrict__ xb, const ushort_t* __restrict__ w1b,
    const float* __restrict__ b1, const int* __restrict__ slot_token,
    const int* __restrict__ bp, ushort_t* __restrict__ H1)
{
    __shared__ __align__(16) ushort_t As[2][128 * 40];
    __shared__ __align__(16) ushort_t Bs[2][128 * 32];
    __shared__ __align__(16) ushort_t Cs[64 * 136];
    __shared__ int toks[128];

    int start = blockIdx.x * 128;
    if (start >= bp[E_]) return;
    int e = 0;
    while (e < E_ - 1 && start >= bp[e + 1]) ++e;
    int n0 = blockIdx.y * 128;
    int tid = threadIdx.x;
    if (tid < 128) toks[tid] = slot_token[start + tid];
    __syncthreads();

    const ushort_t* wB = w1b + (size_t)e * D_ * F_ + n0;

    int myM = tid >> 1;
    int myK = (tid & 1) * 16;
    int myTok = toks[myM];
    // pad rows (tok<0) read token 0: their H1 garbage only reaches pad output
    // rows which the GEMM2 scatter skips.
    const uint4* xr = (const uint4*)(xb + (size_t)(myTok >= 0 ? myTok : 0) * D_ + myK);

    int bk0 = tid >> 4;
    int bn0 = (tid & 15) * 8;

    v8f acc[8] = {};
    int wave = tid >> 5, lane = tid & 31, half = lane >> 4, row = lane & 15;
    int aoff = (wave * 16 + row) * 40 + 8 * half;

    uint4 a0 = xr[0], a1 = xr[1];
    uint4 br0 = *(const uint4*)&wB[(size_t)bk0 * F_ + bn0];
    uint4 br1 = *(const uint4*)&wB[(size_t)(bk0 + 16) * F_ + bn0];

    auto stage = [&](int p) {
        *(uint4*)&As[p][myM * 40 + myK]     = a0;
        *(uint4*)&As[p][myM * 40 + myK + 8] = a1;
        PackU bb0, bb1; bb0.q = br0; bb1.q = br1;
#pragma unroll
        for (int j = 0; j < 8; ++j) {
            Bs[p][(bn0 + j) * 32 + bk0]      = bb0.s[j];
            Bs[p][(bn0 + j) * 32 + bk0 + 16] = bb1.s[j];
        }
    };
    stage(0);

    const int NSTEP = D_ / 32;
    for (int s = 0; s < NSTEP; ++s) {
        int p = s & 1;
        __syncthreads();
        if (s + 1 < NSTEP) {
            int k1 = (s + 1) * 32;
            const uint4* nx = xr + (k1 >> 3);
            a0 = nx[0]; a1 = nx[1];
            br0 = *(const uint4*)&wB[(size_t)(k1 + bk0) * F_ + bn0];
            br1 = *(const uint4*)&wB[(size_t)(k1 + 16 + bk0) * F_ + bn0];
        }
        FragU a;
        a.q[0] = *(const uint4*)&As[p][aoff];
        a.q[1] = *(const uint4*)&As[p][aoff + 16];
#pragma unroll
        for (int nt = 0; nt < 8; ++nt) {
            FragU b;
            b.q[0] = *(const uint4*)&Bs[p][(nt * 16 + row) * 32 + 8 * half];
            b.q[1] = *(const uint4*)&Bs[p][(nt * 16 + row) * 32 + 8 * half + 16];
            acc[nt] = __builtin_amdgcn_wmma_f32_16x16x32_bf16(
                false, a.v, false, b.v, (short)0, acc[nt], false, false);
        }
        if (s + 1 < NSTEP) stage(1 - p);
    }

    // staged epilogue: gelu -> bf16 -> LDS -> coalesced b128 stores (2 halves)
#pragma unroll
    for (int p = 0; p < 2; ++p) {
        if ((wave >> 2) == p) {
#pragma unroll
            for (int nt = 0; nt < 8; ++nt) {
#pragma unroll
                for (int r = 0; r < 8; ++r) {
                    int mloc = wave * 16 + r + 8 * half - 64 * p;
                    int n = n0 + nt * 16 + row;
                    float v = acc[nt][r] + b1[e * F_ + n];
                    Cs[mloc * 136 + nt * 16 + row] = f2bf(gelu_tanh(v));
                }
            }
        }
        __syncthreads();
#pragma unroll
        for (int i = 0; i < 4; ++i) {
            int lin = i * 256 + tid;        // 1024 uint4 chunks = 64 rows
            int m = lin >> 4, c = lin & 15;
            uint4 q = *(const uint4*)&Cs[m * 136 + c * 8];
            *(uint4*)&H1[(size_t)(start + 64 * p + m) * F_ + n0 + c * 8] = q;
        }
        __syncthreads();
    }
}

// ---------------- K6: grouped GEMM2 + weighted scatter-add ----------------
__global__ __launch_bounds__(256) void moxe_ffn2(
    const ushort_t* __restrict__ H1, const ushort_t* __restrict__ w2b,
    const float* __restrict__ b2, const int* __restrict__ slot_token,
    const float* __restrict__ slot_weight, const int* __restrict__ bp,
    float* __restrict__ outF)
{
    __shared__ __align__(16) ushort_t As[2][128 * 40];
    __shared__ __align__(16) ushort_t Bs[2][128 * 32];
    __shared__ int   toks[128];
    __shared__ float wts[128];

    int start = blockIdx.x * 128;
    if (start >= bp[E_]) return;
    int e = 0;
    while (e < E_ - 1 && start >= bp[e + 1]) ++e;
    int n0 = blockIdx.y * 128;
    int tid = threadIdx.x;
    if (tid < 128) { toks[tid] = slot_token[start + tid]; wts[tid] = slot_weight[start + tid]; }
    __syncthreads();

    const ushort_t* wB = w2b + (size_t)e * F_ * D_ + n0;

    int myM = tid >> 1;
    int myK = (tid & 1) * 16;
    const uint4* hr = (const uint4*)&H1[(size_t)(start + myM) * F_ + myK];

    int bk0 = tid >> 4;
    int bn0 = (tid & 15) * 8;

    v8f acc[8] = {};
    int wave = tid >> 5, lane = tid & 31, half = lane >> 4, row = lane & 15;
    int aoff = (wave * 16 + row) * 40 + 8 * half;

    uint4 a0 = hr[0], a1 = hr[1];
    uint4 br0 = *(const uint4*)&wB[(size_t)bk0 * D_ + bn0];
    uint4 br1 = *(const uint4*)&wB[(size_t)(bk0 + 16) * D_ + bn0];

    auto stage = [&](int p) {
        *(uint4*)&As[p][myM * 40 + myK]     = a0;
        *(uint4*)&As[p][myM * 40 + myK + 8] = a1;
        PackU bb0, bb1; bb0.q = br0; bb1.q = br1;
#pragma unroll
        for (int j = 0; j < 8; ++j) {
            Bs[p][(bn0 + j) * 32 + bk0]      = bb0.s[j];
            Bs[p][(bn0 + j) * 32 + bk0 + 16] = bb1.s[j];
        }
    };
    stage(0);

    const int NSTEP = F_ / 32;
    for (int s = 0; s < NSTEP; ++s) {
        int p = s & 1;
        __syncthreads();
        if (s + 1 < NSTEP) {
            int k1 = (s + 1) * 32;
            const uint4* nh = hr + (k1 >> 3);
            a0 = nh[0]; a1 = nh[1];
            br0 = *(const uint4*)&wB[(size_t)(k1 + bk0) * D_ + bn0];
            br1 = *(const uint4*)&wB[(size_t)(k1 + 16 + bk0) * D_ + bn0];
        }
        FragU a;
        a.q[0] = *(const uint4*)&As[p][aoff];
        a.q[1] = *(const uint4*)&As[p][aoff + 16];
#pragma unroll
        for (int nt = 0; nt < 8; ++nt) {
            FragU b;
            b.q[0] = *(const uint4*)&Bs[p][(nt * 16 + row) * 32 + 8 * half];
            b.q[1] = *(const uint4*)&Bs[p][(nt * 16 + row) * 32 + 8 * half + 16];
            acc[nt] = __builtin_amdgcn_wmma_f32_16x16x32_bf16(
                false, a.v, false, b.v, (short)0, acc[nt], false, false);
        }
        if (s + 1 < NSTEP) stage(1 - p);
    }
#pragma unroll
    for (int nt = 0; nt < 8; ++nt) {
#pragma unroll
        for (int r = 0; r < 8; ++r) {
            int mrow = wave * 16 + r + 8 * half;
            int tok = toks[mrow];
            if (tok >= 0) {
                int n = n0 + nt * 16 + row;
                float y = acc[nt][r] + b2[e * D_ + n];
                atomicAdd(&outF[(size_t)tok * D_ + n], wts[mrow] * y);
            }
        }
    }
}

// ---------------- launch ----------------
extern "C" void kernel_launch(void* const* d_in, const int* in_sizes, int n_in,
                              void* d_out, int out_size, void* d_ws, size_t ws_size,
                              hipStream_t stream) {
    (void)in_sizes; (void)n_in; (void)out_size; (void)ws_size;
    const float* x   = (const float*)d_in[0];
    const float* wg  = (const float*)d_in[1];
    const float* bg  = (const float*)d_in[2];
    const float* w1  = (const float*)d_in[3];
    const float* b1  = (const float*)d_in[4];
    const float* w2  = (const float*)d_in[5];
    const float* b2  = (const float*)d_in[6];
    // top_k (d_in[7]) assumed == 2 per setup_inputs.

    float* out = (float*)d_out;
    float* out_gate  = out;
    float* out_probs = out + (size_t)T_ * E_;
    float* out_fhs   = out + (size_t)2 * T_ * E_;
    float* out_z     = out_fhs + (size_t)T_ * D_;
    float* out_lb    = out_z + 1;
    float* out_el    = out_lb + 1;
    float* out_ec    = out_el + E_;

    char* ws = (char*)d_ws;
    size_t o = 0;
    auto carve = [&](size_t bytes) { char* p = ws + o; o = (o + bytes + 255) & ~(size_t)255; return p; };
    ushort_t* w1b  = (ushort_t*)carve((size_t)E_ * D_ * F_ * 2);
    ushort_t* w2b  = (ushort_t*)carve((size_t)E_ * F_ * D_ * 2);
    ushort_t* xb   = (ushort_t*)carve((size_t)T_ * D_ * 2);
    ushort_t* H1   = (ushort_t*)carve((size_t)SLOT_CAP * F_ * 2);
    int*   slot_token  = (int*)carve(SLOT_CAP * 4);
    float* slot_weight = (float*)carve(SLOT_CAP * 4);
    int*   top_idx = (int*)carve(T_ * 2 * 4);
    float* top_w   = (float*)carve(T_ * 2 * 4);
    int*   cnt     = (int*)carve(E_ * 4);
    int*   fill    = (int*)carve(E_ * 4);
    int*   bp      = (int*)carve((E_ + 1) * 4);
    float* probsum = (float*)carve(E_ * 4);
    float* zsum    = (float*)carve(4);

    moxe_init<<<(T_ * D_ + 255) / 256, 256, 0, stream>>>(out_fhs, slot_token, cnt, fill, probsum, zsum);
    moxe_cvt<<<8192, 256, 0, stream>>>(w1, w2, x, w1b, w2b, xb);
    moxe_router<<<T_ / 8, 256, 0, stream>>>(x, wg, bg, out_gate, out_probs,
                                            top_idx, top_w, cnt, probsum, zsum);
    moxe_finalize<<<1, 32, 0, stream>>>(cnt, probsum, zsum, bp, out_z, out_lb, out_el, out_ec);
    moxe_assign<<<(T_ + 255) / 256, 256, 0, stream>>>(top_idx, top_w, bp, fill, slot_token, slot_weight);
    moxe_ffn1<<<dim3(MTILES, F_ / 128), 256, 0, stream>>>(xb, w1b, b1, slot_token, bp, H1);
    moxe_ffn2<<<dim3(MTILES, D_ / 128), 256, 0, stream>>>(H1, w2b, b2, slot_token, slot_weight, bp, out_fhs);
}
